// MHA_10084583211124
// MI455X (gfx1250) — matrix operand order repaired and checked by reading, not compiled, not run on hardware
//
#include <hip/hip_runtime.h>

#define D_MODEL 512
#define NHEAD   8
#define DK      64
#define BATCH   2
#define SEQ     2048

typedef __attribute__((ext_vector_type(16))) _Float16 v16h;
typedef __attribute__((ext_vector_type(8)))  float    v8f;

__device__ __forceinline__ v8f wmma_f16(v16h a, v16h b, v8f c) {
  // D = A(16x32,f16) * B(32x16,f16) + C(16x16,f32)
  return __builtin_amdgcn_wmma_f32_16x16x32_f16(false, a, false, b, (short)0, c,
                                                false, false);
}

// CDNA5 async global->LDS copy (ASYNCcnt path).  Each active lane copies 16B:
//   LDS[lds_off] = MEM[saddr + vaddr]
__device__ __forceinline__ void async_copy_b128(unsigned lds_off, unsigned voff,
                                                unsigned long long base) {
  asm volatile("global_load_async_to_lds_b128 %0, %1, %2"
               :: "v"(lds_off), "v"(voff), "s"(base)
               : "memory");
}
__device__ __forceinline__ void wait_async_le4() {
  asm volatile("s_wait_asynccnt 0x4" ::: "memory");
}
__device__ __forceinline__ void wait_async_le0() {
  asm volatile("s_wait_asynccnt 0x0" ::: "memory");
}

// ---------------------------------------------------------------------------
// Kernel 1: per-head projections.  qh[b,h,n,dk], kh[b,h,n,dk] row-major f16,
// vhT[b,h,dk,n] (transposed) f16.  One wave computes one 16x16 tile.
// ---------------------------------------------------------------------------
__global__ __launch_bounds__(128) void proj_kernel(
    const float* __restrict__ q, const float* __restrict__ k,
    const float* __restrict__ v, const float* __restrict__ qp,
    const float* __restrict__ kp, const float* __restrict__ vp,
    _Float16* __restrict__ qh, _Float16* __restrict__ kh,
    _Float16* __restrict__ vhT) {
  const int lane = threadIdx.x & 31;
  const int wave = threadIdx.x >> 5;
  int gw = blockIdx.x * 4 + wave;  // ((t*8 + h)*256 + mtile)*4 + ntile
  const int ntile = gw & 3;   gw >>= 2;
  const int mtile = gw & 255; gw >>= 8;
  const int h     = gw & 7;   gw >>= 3;
  const int t     = gw;
  if (t >= 3) return;

  const float* X = (t == 0) ? q : (t == 1) ? k : v;
  const float* P = ((t == 0) ? qp : (t == 1) ? kp : vp) + (size_t)h * D_MODEL * DK;

  const int half = lane >> 4;
  const int l16  = lane & 15;
  const int mrow = mtile * 16 + l16;  // A-matrix row (M = lane%16)
  const int nb   = ntile * 16;

  v8f acc = 0.0f;
  for (int kb = 0; kb < D_MODEL; kb += 32) {
    // A fragment: lane row = l16; K = kb + half*16 + {0..15}
    v16h a;
    const float* xr = X + (size_t)mrow * D_MODEL + kb + half * 16;
#pragma unroll
    for (int e = 0; e < 8; ++e) { a[e] = (_Float16)xr[e]; a[8 + e] = (_Float16)xr[8 + e]; }
    // B fragment: element (kdim,n): n = nb+l16, kdim = kb + half*16 + e
    v16h bf;
    const float* pr = P + (size_t)(kb + half * 16) * DK + (nb + l16);
#pragma unroll
    for (int e = 0; e < 16; ++e) bf[e] = (_Float16)pr[e * DK];
    acc = wmma_f16(a, bf, acc);
  }
  // Branch-free store: select base + strides once.  A 16-row tile never
  // crosses the batch boundary (SEQ multiple of 16).
  const int bb  = (mtile * 16) >> 11;
  const int nn0 = (mtile * 16) & (SEQ - 1);
  _Float16* base;
  int sN, sD;  // element strides for (seq-row, d-col)
  if (t == 0)      { base = qh  + (((size_t)bb * NHEAD + h) * SEQ) * DK; sN = DK; sD = 1; }
  else if (t == 1) { base = kh  + (((size_t)bb * NHEAD + h) * SEQ) * DK; sN = DK; sD = 1; }
  else             { base = vhT + (((size_t)bb * NHEAD + h) * DK) * SEQ; sN = 1;  sD = SEQ; }
  // D layout: lane,reg r -> M = r + 8*half, N = l16
#pragma unroll
  for (int r = 0; r < 8; ++r) {
    base[(size_t)(nn0 + r + 8 * half) * sN + (size_t)(nb + l16) * sD] = (_Float16)acc[r];
  }
}

// ---------------------------------------------------------------------------
// Kernel 2: flash-style attention with RBF distance modulation.
// Block = 4 waves sharing one (b,h); each wave owns 16 query rows (64/block).
// K/V tiles (32 keys) are double-buffered in LDS via async global->LDS B128
// copies; each wave stages a quarter (4 async instrs -> s_wait_asynccnt 0x4).
// ---------------------------------------------------------------------------
__global__ __launch_bounds__(128) void attn_kernel(
    const _Float16* __restrict__ qh, const _Float16* __restrict__ kh,
    const _Float16* __restrict__ vhT, const float* __restrict__ coords,
    const unsigned char* __restrict__ pad, _Float16* __restrict__ ctx) {
  __shared__ _Float16 ldsK[2][32 * 64];   // [buf][key_local*64 + d]      4KB/buf
  __shared__ _Float16 ldsV[2][64 * 32];   // [buf][d*32 + key_local]      4KB/buf
  __shared__ _Float16 ldsP[4][16][40];    // per-wave P tile, padded rows

  const int lane = threadIdx.x & 31;
  const int wave = threadIdx.x >> 5;
  const int bi = blockIdx.x;            // (b*8 + h)*32 + qblk
  const int qblk = bi & 31;
  const int h = (bi >> 5) & 7;
  const int b = bi >> 8;

  const int half = lane >> 4, l16 = lane & 15;
  const int q0 = qblk * 64 + wave * 16;

  // spread for this head: 3.7 + (20^(0.98*h/7) - 1)/19 * 16.3
  const float xs = 0.98f * (float)h / 7.0f;
  const float sp = 3.7f + (__expf(xs * 2.9957322736f) - 1.0f) * (16.3f / 19.0f);
  const float sp3 = 3.0f * sp;
  const float inv2s2 = 1.0f / (2.0f * sp * sp);
  const float NEG = -__builtin_inff();

  const _Float16* Qb = qh + (((size_t)b * NHEAD + h) * SEQ + q0) * DK;
  const _Float16* Kb = kh + (((size_t)b * NHEAD + h) * SEQ) * DK;
  const _Float16* Vb = vhT + (((size_t)b * NHEAD + h) * DK) * SEQ;
  const unsigned long long KbB = (unsigned long long)(uintptr_t)Kb;
  const unsigned long long VbB = (unsigned long long)(uintptr_t)Vb;
  const unsigned ldsKoff[2] = {(unsigned)(uintptr_t)&ldsK[0][0],
                               (unsigned)(uintptr_t)&ldsK[1][0]};
  const unsigned ldsVoff[2] = {(unsigned)(uintptr_t)&ldsV[0][0],
                               (unsigned)(uintptr_t)&ldsV[1][0]};

  // Q as two A fragments (dchunks 0..31, 32..63)
  v16h aq0, aq1;
  {
    const _Float16* qr = Qb + (size_t)l16 * DK;
#pragma unroll
    for (int e = 0; e < 8; ++e) {
      aq0[e] = qr[half * 16 + e];       aq0[8 + e] = qr[8 + half * 16 + e];
      aq1[e] = qr[32 + half * 16 + e];  aq1[8 + e] = qr[40 + half * 16 + e];
    }
  }
  // per-register-row query coordinates (rows q0 + r + 8*half)
  float qx[8], qy[8], qz[8];
#pragma unroll
  for (int r = 0; r < 8; ++r) {
    const float* c3 = coords + ((size_t)b * SEQ + q0 + r + 8 * half) * 3;
    qx[r] = c3[0]; qy[r] = c3[1]; qz[r] = c3[2];
  }

  float mrun[8], lrun[8];
  v8f o[4];
#pragma unroll
  for (int r = 0; r < 8; ++r) { mrun[r] = NEG; lrun[r] = 0.0f; }
#pragma unroll
  for (int j = 0; j < 4; ++j) o[j] = 0.0f;

  // ---- async stage of one 32-key tile (this wave's quarter: 4 instrs) ----
  auto stage = [&](int kt, int buf) {
    // K tile: 4KB contiguous (32 rows x 128B).  Wave copies 1KB as 2x512B.
    {
      const unsigned long long sb = KbB + (unsigned)kt * 4096u + (unsigned)wave * 1024u;
      const unsigned kv = (unsigned)lane * 16u;
      const unsigned ld = ldsKoff[buf] + (unsigned)wave * 1024u + kv;
      async_copy_b128(ld, kv, sb);
      async_copy_b128(ld + 512u, kv + 512u, sb);
    }
    // V tile: 64 rows (d) x 64B, global row stride SEQ*2 bytes.
    // Wave copies d rows [wave*16, wave*16+16): 2 instrs x (8 rows x 4 lanes).
    {
      const unsigned long long sb = VbB + (unsigned)kt * 64u;
      const unsigned c = ((unsigned)lane & 3u) * 16u;
#pragma unroll
      for (int i = 0; i < 2; ++i) {
        const unsigned d = (unsigned)wave * 16u + (unsigned)i * 8u + ((unsigned)lane >> 2);
        async_copy_b128(ldsVoff[buf] + d * 64u + c, d * (SEQ * 2u) + c, sb);
      }
    }
  };

  stage(0, 0);
  const int NT = SEQ / 32;
  for (int kt = 0; kt < NT; ++kt) {
    const int cur = kt & 1;
    if (kt + 1 < NT) { stage(kt + 1, cur ^ 1); wait_async_le4(); }
    else             { wait_async_le0(); }
    __syncthreads();                       // all waves' quarters visible
    const _Float16* Ktile = &ldsK[cur][0];
    const _Float16* Vtile = &ldsV[cur][0];
    const int kb2 = kt * 32;

    // --- S = Q K^T for two 16-key tiles (from LDS) ---
    v8f s[2];
#pragma unroll
    for (int nt = 0; nt < 2; ++nt) {
      const _Float16* kr = Ktile + (size_t)(nt * 16 + l16) * 64 + half * 16;
      v16h b0, b1;
#pragma unroll
      for (int e = 0; e < 16; ++e) { b0[e] = kr[e]; b1[e] = kr[32 + e]; }
      v8f c = 0.0f;
      c = wmma_f16(aq0, b0, c);
      c = wmma_f16(aq1, b1, c);
      s[nt] = c;
    }
    // key coords + padding mask (one key per lane per n-tile)
    float kx[2], ky[2], kz[2]; int km[2];
#pragma unroll
    for (int nt = 0; nt < 2; ++nt) {
      const int key = kb2 + nt * 16 + l16;
      const float* c3 = coords + ((size_t)b * SEQ + key) * 3;
      kx[nt] = c3[0]; ky[nt] = c3[1]; kz[nt] = c3[2];
      km[nt] = pad[(size_t)b * SEQ + key] ? 1 : 0;
    }
    // --- RBF modulation + mask ---
    float rmax[8];
#pragma unroll
    for (int r = 0; r < 8; ++r) rmax[r] = NEG;
#pragma unroll
    for (int nt = 0; nt < 2; ++nt) {
#pragma unroll
      for (int r = 0; r < 8; ++r) {
        float attnv = s[nt][r] * 0.125f;  // 1/sqrt(64)
        const float dx = qx[r] - kx[nt], dy = qy[r] - ky[nt], dz = qz[r] - kz[nt];
        float d = sqrtf(dx * dx + dy * dy + dz * dz);
        d = fmaxf(d, sp);
        const bool msk = (d > sp3) || km[nt];
        d = fminf(d, sp3);
        const float arg = d * d * inv2s2;
        const float ef = __expf(attnv < 0.0f ? arg : -arg);  // 1/rbf if attn<0
        const float val = msk ? NEG : attnv * ef;
        s[nt][r] = val;
        rmax[r] = fmaxf(rmax[r], val);
      }
    }
    // --- online softmax: row reductions inside 16-lane halves (wave32) ---
#pragma unroll
    for (int r = 0; r < 8; ++r) {
      float v = rmax[r];
      v = fmaxf(v, __shfl_xor(v, 1, 32));
      v = fmaxf(v, __shfl_xor(v, 2, 32));
      v = fmaxf(v, __shfl_xor(v, 4, 32));
      v = fmaxf(v, __shfl_xor(v, 8, 32));
      const float mnew = fmaxf(mrun[r], v);
      const bool fin = (mnew > -1e30f);
      const float corr = fin ? __expf(mrun[r] - mnew) : 1.0f;
      mrun[r] = mnew;
      float ps = 0.0f;
#pragma unroll
      for (int nt = 0; nt < 2; ++nt) {
        const float p = fin ? __expf(s[nt][r] - mnew) : 0.0f;
        s[nt][r] = p;
        ps += p;
      }
      ps += __shfl_xor(ps, 1, 32);
      ps += __shfl_xor(ps, 2, 32);
      ps += __shfl_xor(ps, 4, 32);
      ps += __shfl_xor(ps, 8, 32);
      lrun[r] = lrun[r] * corr + ps;
#pragma unroll
      for (int j = 0; j < 4; ++j) o[j][r] *= corr;
    }
    // --- P: C-layout -> A-layout via per-wave LDS tile ---
    __syncthreads();
#pragma unroll
    for (int r = 0; r < 8; ++r) {
      ldsP[wave][r + 8 * half][l16]      = (_Float16)s[0][r];
      ldsP[wave][r + 8 * half][16 + l16] = (_Float16)s[1][r];
    }
    __syncthreads();
    v16h ap;
    {
      const _Float16* pr = &ldsP[wave][l16][0];
#pragma unroll
      for (int e = 0; e < 8; ++e) {
        ap[e] = pr[half * 16 + e];
        ap[8 + e] = pr[8 + half * 16 + e];
      }
    }
    // --- O += P V (V from LDS) ---
#pragma unroll
    for (int j = 0; j < 4; ++j) {
      const _Float16* vr = Vtile + (size_t)(j * 16 + l16) * 32 + half * 16;
      v16h bv;
#pragma unroll
      for (int e = 0; e < 16; ++e) bv[e] = vr[e];
      o[j] = wmma_f16(ap, bv, o[j]);
    }
    __syncthreads();   // protect both LDS buffers before next stage overwrites
  }
  // --- normalize and write ctx[b, n, h*64 + dk] ---
#pragma unroll
  for (int r = 0; r < 8; ++r) {
    const float inv = (lrun[r] > 0.0f) ? 1.0f / lrun[r] : 0.0f;
    _Float16* cr = ctx + ((size_t)b * SEQ + q0 + r + 8 * half) * D_MODEL + h * DK;
#pragma unroll
    for (int j = 0; j < 4; ++j) cr[j * 16 + l16] = (_Float16)(o[j][r] * inv);
  }
}

// ---------------------------------------------------------------------------
// Kernel 3: Y = ctx @ W^T + bias (fp32 out).  One wave per 16x16 tile.
// ---------------------------------------------------------------------------
__global__ __launch_bounds__(128) void outproj_kernel(
    const _Float16* __restrict__ ctx, const float* __restrict__ W,
    const float* __restrict__ bias, float* __restrict__ out) {
  const int lane = threadIdx.x & 31;
  const int wave = threadIdx.x >> 5;
  const int gw = blockIdx.x * 4 + wave;
  const int ntile = gw & 31;
  const int mtile = gw >> 5;
  const int half = lane >> 4, l16 = lane & 15;
  const int mrow = mtile * 16 + l16;

  v8f acc = 0.0f;
  for (int kb = 0; kb < D_MODEL; kb += 32) {
    v16h a;
    const _Float16* cr = ctx + (size_t)mrow * D_MODEL + kb + half * 16;
#pragma unroll
    for (int e = 0; e < 8; ++e) { a[e] = cr[e]; a[8 + e] = cr[8 + e]; }
    // B element (k,n) = W[n][k]: n = ntile*16+l16, k = kb + half*16 + e
    v16h bf;
    const float* wr = W + (size_t)(ntile * 16 + l16) * D_MODEL + kb + half * 16;
#pragma unroll
    for (int e = 0; e < 16; ++e) bf[e] = (_Float16)wr[e];
    acc = wmma_f16(a, bf, acc);
  }
#pragma unroll
  for (int r = 0; r < 8; ++r) {
    const int g = mtile * 16 + r + 8 * half;
    const int n = ntile * 16 + l16;
    out[(size_t)g * D_MODEL + n] = acc[r] + bias[n];
  }
}

// ---------------------------------------------------------------------------
extern "C" void kernel_launch(void* const* d_in, const int* in_sizes, int n_in,
                              void* d_out, int out_size, void* d_ws, size_t ws_size,
                              hipStream_t stream) {
  const float* q  = (const float*)d_in[0];
  const float* k  = (const float*)d_in[1];
  const float* v  = (const float*)d_in[2];
  const float* co = (const float*)d_in[3];
  const unsigned char* pad = (const unsigned char*)d_in[4];
  const float* qp = (const float*)d_in[5];
  const float* kp = (const float*)d_in[6];
  const float* vp = (const float*)d_in[7];
  const float* W  = (const float*)d_in[8];
  const float* bi = (const float*)d_in[9];

  const size_t perhead = (size_t)BATCH * NHEAD * SEQ * DK;  // 2,097,152 elems
  _Float16* qh  = (_Float16*)d_ws;
  _Float16* kh  = qh + perhead;
  _Float16* vhT = kh + perhead;
  _Float16* ctx = vhT + perhead;                            // total 16 MB

  // projections: 3 tensors * 8 heads * 256 mtiles * 4 ntiles = 24576 waves
  proj_kernel<<<6144, 128, 0, stream>>>(q, k, v, qp, kp, vp, qh, kh, vhT);
  // attention: B*H*(SEQ/64) blocks, 4 waves each sharing LDS-staged K/V
  attn_kernel<<<512, 128, 0, stream>>>(qh, kh, vhT, co, pad, ctx);
  // output projection: 256 mtiles * 32 ntiles = 8192 waves
  outproj_kernel<<<2048, 128, 0, stream>>>(ctx, W, bi, (float*)d_out);
  (void)in_sizes; (void)n_in; (void)out_size; (void)ws_size;
}
